// JMP_9363028705439
// MI455X (gfx1250) — compile-verified, matrix operand order checked by reading, not run on hardware
//
#include <hip/hip_runtime.h>
#include <stdint.h>

#define N_ATOMS 4096
#define NG      12
#define N_GRID  (NG*NG*NG)
#define N_TOT   (N_ATOMS + N_GRID)
#define HID     256
#define KNN     32
#define NRBF    64
#define CUTOFF  12.0f
#define REPULSION 2.0f
#define VNODE_Z 120

typedef __attribute__((ext_vector_type(16))) __bf16 v16bf;
typedef __attribute__((ext_vector_type(8)))  float  v8f;

struct U16x2 { uint4 lo, hi; };

static __device__ inline v16bf mk16(uint4 lo, uint4 hi) {
  U16x2 u{lo, hi};
  return __builtin_bit_cast(v16bf, u);
}

// fp32 -> bf16 round-to-nearest-even, returned as raw bits
static __device__ inline uint16_t f2bf(float x) {
  uint32_t u = __float_as_uint(x);
  u += 0x7FFFu + ((u >> 16) & 1u);
  return (uint16_t)(u >> 16);
}

// ---- WMMA tile loaders (ISA 7.12.2 layouts, wave32) -------------------------
// A 16x32 bf16: lane -> M=lane%16; K pattern per VGPR v: K = (v%4)*2+(v/4)*16
// + 8*(lane/16) + (elem&1).  With base = tile(row0,k0), per lane this is two
// contiguous 16-byte chunks at +0 (K 0..7) and +16 elems (K 16..23), shifted
// by 8 for the upper lane half -> 2x (ds_load_)b128.
static __device__ inline v16bf load_a_lds(const uint16_t* base, int stride) {
  int lane = threadIdx.x & 31;
  const uint16_t* p = base + (lane & 15) * stride + ((lane >> 4) << 3);
  uint4 lo = *(const uint4*)p;
  uint4 hi = *(const uint4*)(p + 16);
  return mk16(lo, hi);
}
// B 32x16 bf16 from an N-major (K-contiguous) transposed weight matrix:
// lane -> N = n0 + lane%16; same K packing as A -> 2x global_load_b128.
static __device__ inline v16bf load_b_gbl(const uint16_t* __restrict__ WT,
                                          int n0, int Kdim, int k0) {
  int lane = threadIdx.x & 31;
  const uint16_t* p = WT + (size_t)(n0 + (lane & 15)) * Kdim + k0 + ((lane >> 4) << 3);
  uint4 lo = *(const uint4*)p;
  uint4 hi = *(const uint4*)(p + 16);
  return mk16(lo, hi);
}

static __device__ inline v8f wmma_bf(v16bf a, v16bf b, v8f c) {
  // (neg_a, A, neg_b, B, c_mod, C, reuse_a, reuse_b)
  return __builtin_amdgcn_wmma_f32_16x16x32_bf16(false, a, false, b, (short)0, c,
                                                 false, false);
}

// ---- kernel 1: copy atoms into all_pos, valid=1 -----------------------------
__global__ void k_init_atoms(const float* __restrict__ pos,
                             float* __restrict__ allpos, int* __restrict__ valid) {
  int i = blockIdx.x * blockDim.x + threadIdx.x;
  if (i < N_ATOMS) {
    allpos[i * 3 + 0] = pos[i * 3 + 0];
    allpos[i * 3 + 1] = pos[i * 3 + 1];
    allpos[i * 3 + 2] = pos[i * 3 + 2];
    valid[i] = 1;
  }
}

// ---- kernel 2: grid node positions + repulsion mask -------------------------
__global__ void __launch_bounds__(256) k_grid(const float* __restrict__ pos,
                                              const float* __restrict__ cell,
                                              float* __restrict__ allpos,
                                              int* __restrict__ valid) {
  __shared__ float red[256];
  int g = blockIdx.x;
  int ix = g / (NG * NG), iy = (g / NG) % NG, iz = g % NG;
  float fx = (float)ix / NG, fy = (float)iy / NG, fz = (float)iz / NG;
  float gx = fx * cell[0] + fy * cell[3] + fz * cell[6];
  float gy = fx * cell[1] + fy * cell[4] + fz * cell[7];
  float gz = fx * cell[2] + fy * cell[5] + fz * cell[8];
  float mn = 3.4e38f;
  for (int a = threadIdx.x; a < N_ATOMS; a += 256) {
    float dx = pos[a * 3 + 0] - gx;
    float dy = pos[a * 3 + 1] - gy;
    float dz = pos[a * 3 + 2] - gz;
    float d2 = dx * dx + dy * dy + dz * dz;
    mn = fminf(mn, d2);
  }
  red[threadIdx.x] = mn;
  __syncthreads();
  for (int off = 128; off > 0; off >>= 1) {
    if ((int)threadIdx.x < off) red[threadIdx.x] = fminf(red[threadIdx.x], red[threadIdx.x + off]);
    __syncthreads();
  }
  if (threadIdx.x == 0) {
    int i = N_ATOMS + g;
    allpos[i * 3 + 0] = gx;
    allpos[i * 3 + 1] = gy;
    allpos[i * 3 + 2] = gz;
    valid[i] = (sqrtf(fmaxf(red[0], 1e-12f)) >= REPULSION) ? 1 : 0;
  }
}

// ---- kernel 3: transpose+convert weights to bf16, N-major -------------------
__global__ void k_convw(const float* __restrict__ Wrbf, const float* __restrict__ W1,
                        const float* __restrict__ W2, uint16_t* __restrict__ WrbfT,
                        uint16_t* __restrict__ W1T, uint16_t* __restrict__ W2T) {
  int t = blockIdx.x * 256 + threadIdx.x;
  if (t < NRBF * HID) {
    int k = t / HID, n = t % HID;
    WrbfT[n * NRBF + k] = f2bf(Wrbf[t]);
  } else if (t < NRBF * HID + HID * HID) {
    int t1 = t - NRBF * HID;
    int k = t1 / HID, n = t1 % HID;
    W1T[n * HID + k] = f2bf(W1[t1]);
  } else if (t < NRBF * HID + 2 * HID * HID) {
    int t2 = t - (NRBF * HID + HID * HID);
    int k = t2 / HID, n = t2 % HID;
    W2T[n * HID + k] = f2bf(W2[t2]);
  }
}

// ---- kernel 4: KNN (K=32 nearest) per node ----------------------------------
__global__ void __launch_bounds__(256) k_knn(const float* __restrict__ allpos,
                                             const int* __restrict__ valid,
                                             int* __restrict__ nbr,
                                             float* __restrict__ dedge) {
  __shared__ float d2[N_TOT];        // 23.3 KB
  __shared__ float rv[256];
  __shared__ int   ri[256];
  int i = blockIdx.x;
  int tid = threadIdx.x;
  float px = allpos[i * 3 + 0], py = allpos[i * 3 + 1], pz = allpos[i * 3 + 2];
  int vi = valid[i];
  for (int j = tid; j < N_TOT; j += 256) {
    float dx = allpos[j * 3 + 0] - px;
    float dy = allpos[j * 3 + 1] - py;
    float dz = allpos[j * 3 + 2] - pz;
    float dd = dx * dx + dy * dy + dz * dz;
    if (j == i || vi == 0 || valid[j] == 0) dd = 1e18f;  // sqrt(1e18)=1e9 == BIG
    d2[j] = dd;
  }
  __syncthreads();
  for (int s = 0; s < KNN; ++s) {
    float bv = 3.4e38f;
    int bi = 0x7FFFFFFF;
    for (int j = tid; j < N_TOT; j += 256) {
      float v = d2[j];
      if (v < bv || (v == bv && j < bi)) { bv = v; bi = j; }
    }
    rv[tid] = bv; ri[tid] = bi;
    __syncthreads();
    for (int off = 128; off > 0; off >>= 1) {
      if (tid < off) {
        float ov = rv[tid + off]; int oi = ri[tid + off];
        if (ov < rv[tid] || (ov == rv[tid] && oi < ri[tid])) { rv[tid] = ov; ri[tid] = oi; }
      }
      __syncthreads();
    }
    if (tid == 0) {
      int sel = ri[0];
      nbr[i * KNN + s] = sel;
      dedge[i * KNN + s] = sqrtf(fmaxf(rv[0], 1e-12f));
      d2[sel] = 3.0e38f;               // exclude from later passes
    }
    __syncthreads();
  }
}

// ---- kernel 5: embedding gather (h) -----------------------------------------
__global__ void k_embed(const int* __restrict__ anum, const int* __restrict__ valid,
                        const float* __restrict__ embed, float* __restrict__ h) {
  int i = blockIdx.x;
  int c = threadIdx.x;
  int z = (i < N_ATOMS) ? anum[i] : VNODE_Z;
  h[(size_t)i * HID + c] = valid[i] ? embed[(size_t)z * HID + c] : 0.0f;
}

// ---- kernel 6: output masks -------------------------------------------------
__global__ void k_masks(const int* __restrict__ valid, float* __restrict__ vn,
                        float* __restrict__ pn) {
  int i = blockIdx.x * blockDim.x + threadIdx.x;
  if (i < N_TOT) {
    vn[i] = (i >= N_ATOMS && valid[i]) ? 1.0f : 0.0f;
    pn[i] = (i < N_ATOMS) ? 1.0f : 0.0f;
  }
}

// ---- kernel 7: fused edge MLP + aggregation (WMMA bf16) ---------------------
// One workgroup (4 waves) per destination node. 32 edges = 2 M-tiles of 16.
// Wave w owns output columns [64w, 64w+64) = 4 N-tiles.
__global__ void __launch_bounds__(128, 1) k_edge_mlp(
    const float* __restrict__ h, const int* __restrict__ nbr,
    const float* __restrict__ dedge, const int* __restrict__ valid,
    const uint16_t* __restrict__ WrbfT, const uint16_t* __restrict__ W1T,
    const uint16_t* __restrict__ W2T, const float* __restrict__ b1,
    const float* __restrict__ b2, float* __restrict__ feat) {
  __shared__ __align__(16) uint16_t rbfS[KNN * NRBF];  // 4 KB  (bf16 A, stage A)
  __shared__ __align__(16) uint16_t m0S[KNN * HID];    // 16 KB (bf16 A, stage B)
  __shared__ __align__(16) uint16_t tS[KNN * HID];     // 16 KB (bf16 A, stage C)
  __shared__ __align__(16) float hnbrS[KNN * HID];     // 32 KB (staged h[nbr])
  __shared__ float hS[HID];
  __shared__ float dS[KNN];
  __shared__ int   nbrS[KNN];
  __shared__ float maskS[KNN];

  const int i = blockIdx.x;
  const int tid = threadIdx.x;
  const int lane = tid & 31;
  const int wave = tid >> 5;
  const int n0 = wave * 64;
  const int nlane = lane & 15;
  const int mbase = (lane >> 4) << 3;   // C tile: M = mbase + r

  // stage 0a: edge metadata + h_i
  if (tid < KNN) {
    nbrS[tid] = nbr[i * KNN + tid];
    float d = dedge[i * KNN + tid];
    dS[tid] = d;
    maskS[tid] = (d <= CUTOFF && valid[i] != 0) ? 1.0f : 0.0f;
  }
  for (int c = tid; c < HID; c += 128) hS[c] = h[(size_t)i * HID + c];
  __syncthreads();

  // stage 0b: coalesced staging of all 32 neighbor rows (b128 loads) + RBF
  for (int t = tid; t < KNN * (HID / 4); t += 128) {
    int e = t >> 6;                 // edge row
    int c4 = (t & 63) << 2;         // column group of 4
    float4 v = *(const float4*)&h[(size_t)nbrS[e] * HID + c4];
    *(float4*)&hnbrS[e * HID + c4] = v;
  }
  const float width = CUTOFF / NRBF;
  const float cstep = CUTOFF / (NRBF - 1);   // linspace(0, CUTOFF, NRBF)
  for (int t = tid; t < KNN * NRBF; t += 128) {
    int e = t >> 6, r = t & 63;
    float x = (dS[e] - (float)r * cstep) / width;
    rbfS[t] = f2bf(__expf(-0.5f * x * x));
  }
  __syncthreads();

  v8f zero = {0.f, 0.f, 0.f, 0.f, 0.f, 0.f, 0.f, 0.f};
  v8f acc[2][4];

  // ---- stage A: m0 = h_i + h_nbr + rbf @ Wrbf  (M=32,K=64,N=64/wave)
  #pragma unroll
  for (int mt = 0; mt < 2; ++mt)
    #pragma unroll
    for (int nt = 0; nt < 4; ++nt) acc[mt][nt] = zero;
  #pragma unroll
  for (int kt = 0; kt < 2; ++kt) {
    v16bf a0 = load_a_lds(rbfS + kt * 32, NRBF);
    v16bf a1 = load_a_lds(rbfS + 16 * NRBF + kt * 32, NRBF);
    #pragma unroll
    for (int nt = 0; nt < 4; ++nt) {
      v16bf b = load_b_gbl(WrbfT, n0 + nt * 16, NRBF, kt * 32);
      acc[0][nt] = wmma_bf(a0, b, acc[0][nt]);
      acc[1][nt] = wmma_bf(a1, b, acc[1][nt]);
    }
  }
  #pragma unroll
  for (int mt = 0; mt < 2; ++mt)
    #pragma unroll
    for (int nt = 0; nt < 4; ++nt) {
      int col = n0 + nt * 16 + nlane;
      float hv = hS[col];
      #pragma unroll
      for (int r = 0; r < 8; ++r) {
        int row = mt * 16 + mbase + r;
        float v = acc[mt][nt][r] + hv + hnbrS[row * HID + col];
        m0S[row * HID + col] = f2bf(v);
      }
    }
  __syncthreads();

  // ---- stage B: t = silu(m0 @ W1 + b1)  (M=32,K=256,N=64/wave)
  #pragma unroll
  for (int mt = 0; mt < 2; ++mt)
    #pragma unroll
    for (int nt = 0; nt < 4; ++nt) acc[mt][nt] = zero;
  for (int kt = 0; kt < 8; ++kt) {
    v16bf a0 = load_a_lds(m0S + kt * 32, HID);
    v16bf a1 = load_a_lds(m0S + 16 * HID + kt * 32, HID);
    #pragma unroll
    for (int nt = 0; nt < 4; ++nt) {
      v16bf b = load_b_gbl(W1T, n0 + nt * 16, HID, kt * 32);
      acc[0][nt] = wmma_bf(a0, b, acc[0][nt]);
      acc[1][nt] = wmma_bf(a1, b, acc[1][nt]);
    }
  }
  #pragma unroll
  for (int mt = 0; mt < 2; ++mt)
    #pragma unroll
    for (int nt = 0; nt < 4; ++nt) {
      int col = n0 + nt * 16 + nlane;
      float bb = b1[col];
      #pragma unroll
      for (int r = 0; r < 8; ++r) {
        int row = mt * 16 + mbase + r;
        float x = acc[mt][nt][r] + bb;
        float s = x / (1.0f + __expf(-x));       // silu
        tS[row * HID + col] = f2bf(s);
      }
    }
  __syncthreads();

  // ---- stage C: m = t @ W2 + b2, mask, row-sum, + h_i -> feat
  #pragma unroll
  for (int mt = 0; mt < 2; ++mt)
    #pragma unroll
    for (int nt = 0; nt < 4; ++nt) acc[mt][nt] = zero;
  for (int kt = 0; kt < 8; ++kt) {
    v16bf a0 = load_a_lds(tS + kt * 32, HID);
    v16bf a1 = load_a_lds(tS + 16 * HID + kt * 32, HID);
    #pragma unroll
    for (int nt = 0; nt < 4; ++nt) {
      v16bf b = load_b_gbl(W2T, n0 + nt * 16, HID, kt * 32);
      acc[0][nt] = wmma_bf(a0, b, acc[0][nt]);
      acc[1][nt] = wmma_bf(a1, b, acc[1][nt]);
    }
  }
  #pragma unroll
  for (int nt = 0; nt < 4; ++nt) {
    int col = n0 + nt * 16 + nlane;
    float bb = b2[col];
    float s = 0.0f;
    #pragma unroll
    for (int mt = 0; mt < 2; ++mt) {
      #pragma unroll
      for (int r = 0; r < 8; ++r) {
        int row = mt * 16 + mbase + r;
        s += (acc[mt][nt][r] + bb) * maskS[row];
      }
    }
    s += __shfl_xor(s, 16, 32);      // combine M halves (lanes L <-> L+16)
    if (lane < 16) feat[(size_t)i * HID + col] = hS[col] + s;
  }
}

// ---- host launcher ----------------------------------------------------------
extern "C" void kernel_launch(void* const* d_in, const int* in_sizes, int n_in,
                              void* d_out, int out_size, void* d_ws, size_t ws_size,
                              hipStream_t stream) {
  const float* pos   = (const float*)d_in[0];
  const int*   anum  = (const int*)d_in[1];
  const float* cell  = (const float*)d_in[2];
  const float* embed = (const float*)d_in[3];
  const float* Wrbf  = (const float*)d_in[4];
  const float* W1    = (const float*)d_in[5];
  const float* b1    = (const float*)d_in[6];
  const float* W2    = (const float*)d_in[7];
  const float* b2    = (const float*)d_in[8];
  (void)in_sizes; (void)n_in; (void)out_size; (void)ws_size;

  uintptr_t base = (uintptr_t)d_ws;
  auto take = [&](size_t bytes) {
    uintptr_t p = base;
    base += (bytes + 255) & ~(size_t)255;
    return p;
  };
  float*    allpos = (float*)take((size_t)N_TOT * 3 * sizeof(float));
  int*      valid  = (int*)take((size_t)N_TOT * sizeof(int));
  int*      nbr    = (int*)take((size_t)N_TOT * KNN * sizeof(int));
  float*    dedge  = (float*)take((size_t)N_TOT * KNN * sizeof(float));
  float*    h      = (float*)take((size_t)N_TOT * HID * sizeof(float));
  uint16_t* WrbfT  = (uint16_t*)take((size_t)HID * NRBF * sizeof(uint16_t));
  uint16_t* W1T    = (uint16_t*)take((size_t)HID * HID * sizeof(uint16_t));
  uint16_t* W2T    = (uint16_t*)take((size_t)HID * HID * sizeof(uint16_t));

  float* feat = (float*)d_out;
  float* vn   = feat + (size_t)N_TOT * HID;
  float* pn   = vn + N_TOT;

  k_init_atoms<<<(N_ATOMS + 255) / 256, 256, 0, stream>>>(pos, allpos, valid);
  k_grid<<<N_GRID, 256, 0, stream>>>(pos, cell, allpos, valid);
  k_convw<<<(NRBF * HID + 2 * HID * HID + 255) / 256, 256, 0, stream>>>(
      Wrbf, W1, W2, WrbfT, W1T, W2T);
  k_knn<<<N_TOT, 256, 0, stream>>>(allpos, valid, nbr, dedge);
  k_embed<<<N_TOT, HID, 0, stream>>>(anum, valid, embed, h);
  k_masks<<<(N_TOT + 255) / 256, 256, 0, stream>>>(valid, vn, pn);
  k_edge_mlp<<<N_TOT, 128, 0, stream>>>(h, nbr, dedge, valid, WrbfT, W1T, W2T,
                                        b1, b2, feat);
}
